// NeighborhoodCrossAttention_63763084476807
// MI455X (gfx1250) — matrix operand order repaired and checked by reading, not compiled, not run on hardware
//
#include <hip/hip_runtime.h>
#include <math.h>

// Problem constants (fixed by the reference): B=4, C=640, H=W=64, 3x3 neighborhood.
#define BB 4
#define CCH 640
#define HH 64
#define WW 64
#define CK 64              // channel chunk staged in LDS per iteration
#define WROW 66            // padded LDS row stride in bf16 elements (33 dwords -> conflict-free)
#define NCHUNK (CCH / CK)

typedef __attribute__((ext_vector_type(16))) __bf16        v16bf;
typedef __attribute__((ext_vector_type(8)))  float         v8f;
typedef __attribute__((ext_vector_type(8)))  unsigned int  v8u;

static __device__ __forceinline__ unsigned short f32_to_bf16(float f) {
  unsigned int u = __builtin_bit_cast(unsigned int, f);
  u += 0x7fffu + ((u >> 16) & 1u);           // round-to-nearest-even
  return (unsigned short)(u >> 16);
}

__global__ __launch_bounds__(256)
void nca_row_kernel(const float* __restrict__ Ft, const float* __restrict__ Fwp,
                    const float* __restrict__ mask, const float* __restrict__ cw,
                    const float* __restrict__ cb, float* __restrict__ out)
{
  // LDS: bf16 staging for WMMA + f32 attention / reduction scratch (~37 KB total)
  __shared__ unsigned short ft_s[WW * WROW];        // Ft row, [w][c] bf16
  __shared__ unsigned short fw_s[3 * WW * WROW];    // 3 Fwarp rows, [dy][w][c] bf16
  __shared__ float sim_s[WW][12];                   // sim logits, then attn weights
  __shared__ float lamacc_s[WW];
  __shared__ float lam_s[WW];

  const int tid  = threadIdx.x;
  const int lane = tid & 31;
  const int wave = tid >> 5;
  const int bh   = blockIdx.x;
  const int b    = bh / HH;
  const int h    = bh % HH;

  const int hr0 = (h > 0) ? h - 1 : 0;
  const int hr1 = h;
  const int hr2 = (h < HH - 1) ? h + 1 : HH - 1;

  const int m  = lane & 15;    // WMMA row (A) / col (B) index within tile
  const int hi = lane >> 4;    // lane half selector for K striping

  // ---- Chunk-invariant tile decode, hoisted out of the hot loop (no tables) ----
  // Band tiles per dy: r in 0..9 -> qt=(r+1)/3, kt=qt+((r+1)%3)-1
  // gives (0,0),(0,1),(1,0),(1,1),(1,2),(2,1),(2,2),(2,3),(3,2),(3,3).
  bool tact[4];
  int  tdy[4], tqt[4], tkt[4], aoff[4], boff[4];
  #pragma unroll
  for (int s = 0; s < 4; ++s) {
    int t = wave + s * 8;
    tact[s] = (t < 30);
    int tt = tact[s] ? t : 0;
    int dy = tt / 10;
    int r  = tt - dy * 10;
    int qt = (r + 1) / 3;
    int kt = qt + ((r + 1) % 3) - 1;
    tdy[s] = dy; tqt[s] = qt; tkt[s] = kt;
    aoff[s] = (qt * 16 + m) * WROW;                 // Ft fragment row base (half idx)
    boff[s] = (dy * WW + kt * 16 + m) * WROW;       // Fwarp fragment row base
  }

  v8f acc0 = (v8f){0.f,0.f,0.f,0.f,0.f,0.f,0.f,0.f};
  v8f acc1 = acc0, acc2 = acc0, acc3 = acc0;
  v8f* accp[4] = {&acc0, &acc1, &acc2, &acc3};

  // Per-VGPR K offsets (constant): A lanes0-15 K=0..15, lanes16-31 K=16..31;
  // B lanes0-15 K=0..15, lanes16-31 K=16..31 (2 bf16 per VGPR).
  const int kaBase = hi * 16;
  const int kbBase = hi * 16;

  // ------------- Phase A: banded gram via v_wmma_f32_16x16x32_bf16 -------------
  for (int ch = 0; ch < NCHUNK; ++ch) {
    const int c0 = ch * CK;
    __syncthreads();   // previous chunk's WMMA reads done before LDS overwrite

    // stage Ft chunk: consecutive lanes -> consecutive w (coalesced 128B)
    const float* ftg = Ft + ((size_t)(b * CCH + c0) * HH + h) * WW;
    #pragma unroll
    for (int j = 0; j < (CK * WW) / 256; ++j) {
      int i = tid + j * 256;
      int ci = i >> 6, w = i & 63;
      ft_s[w * WROW + ci] = f32_to_bf16(ftg[(size_t)ci * (HH * WW) + w]);
    }
    // stage 3 clamped Fwarp rows
    #pragma unroll
    for (int j = 0; j < (3 * CK * WW) / 256; ++j) {
      int i = tid + j * 256;
      int dy = i / (CK * WW);
      int r  = i - dy * (CK * WW);
      int ci = r >> 6, w = r & 63;
      int hr = (dy == 0) ? hr0 : ((dy == 1) ? hr1 : hr2);
      const float* g = Fwp + ((size_t)(b * CCH + c0 + ci) * HH + hr) * WW;
      fw_s[(dy * WW + w) * WROW + ci] = f32_to_bf16(g[w]);
    }
    if (ch + 1 < NCHUNK) {
      // CDNA5 global_prefetch_b8 of next chunk's Ft slice
      const float* nft = Ft + ((size_t)(b * CCH + c0 + CK + (tid >> 6)) * HH + h) * WW + (tid & 63);
      __builtin_prefetch(nft, 0, 0);
    }
    __syncthreads();

    // each wave accumulates its band tiles (tact is wave-uniform -> EXEC all-1s in WMMA)
    #pragma unroll
    for (int s = 0; s < 4; ++s) {
      if (!tact[s]) continue;
      const unsigned short* arow = &ft_s[aoff[s]];
      const unsigned short* brow = &fw_s[boff[s]];
      #pragma unroll
      for (int k0 = 0; k0 < CK; k0 += 32) {
        v8u au, bu;
        #pragma unroll
        for (int v = 0; v < 8; ++v) {
          int ka = k0 + kaBase + 2 * v;
          int kb = k0 + kbBase + 2 * v;
          au[v] = *reinterpret_cast<const unsigned int*>(&arow[ka]);
          bu[v] = *reinterpret_cast<const unsigned int*>(&brow[kb]);
        }
        *accp[s] = __builtin_amdgcn_wmma_f32_16x16x32_bf16(
            false, __builtin_bit_cast(v16bf, au),
            false, __builtin_bit_cast(v16bf, bu),
            (short)0, *accp[s], false, false);
      }
    }
  }

  // extract +/-1 band into sim_s (C/D layout: VGPR v -> M=v (+8 for lanes>=16), N=lane&15)
  const float scale = rsqrtf((float)CCH);
  #pragma unroll
  for (int s = 0; s < 4; ++s) {
    if (!tact[s]) continue;
    int dy = tdy[s], qt = tqt[s], kt = tkt[s];
    int n_g = kt * 16 + m;
    #pragma unroll
    for (int v = 0; v < 8; ++v) {
      int m_g = qt * 16 + v + hi * 8;
      int d = n_g - m_g;
      float val = (*accp[s])[v] * scale;
      if (d >= -1 && d <= 1)        sim_s[m_g][dy * 3 + d + 1] = val;
      if (m_g == 0  && n_g == 0)    sim_s[0 ][dy * 3 + 0] = val;   // replicate pad w=-1
      if (m_g == 63 && n_g == 63)   sim_s[63][dy * 3 + 2] = val;   // replicate pad w=64
    }
  }
  __syncthreads();

  // ------------- masked softmax over the 9 neighbors (f32) -------------
  if (tid < WW) {
    const int w  = tid;
    const int wl = (w > 0) ? w - 1 : 0;
    const int wr = (w < WW - 1) ? w + 1 : WW - 1;
    const int wc3[3] = {wl, w, wr};
    const int hr3[3] = {hr0, hr1, hr2};
    const float* mrow = mask + (size_t)b * (HH * WW);
    float sv[9];
    bool anyv = false;
    #pragma unroll
    for (int dy = 0; dy < 3; ++dy)
      #pragma unroll
      for (int dx = 0; dx < 3; ++dx) {
        int n = dy * 3 + dx;
        bool occ = (mrow[hr3[dy] * WW + wc3[dx]] == 1.0f);
        sv[n] = occ ? -1e9f : sim_s[w][n];
        anyv |= !occ;
      }
    if (!anyv) {
      #pragma unroll
      for (int n = 0; n < 9; ++n) sim_s[w][n] = 0.0f;   // fully-occluded row -> zero attn
    } else {
      float mx = sv[0];
      #pragma unroll
      for (int n = 1; n < 9; ++n) mx = fmaxf(mx, sv[n]);
      float e[9], den = 0.f;
      #pragma unroll
      for (int n = 0; n < 9; ++n) { e[n] = expf(sv[n] - mx); den += e[n]; }
      float inv = 1.0f / den;
      #pragma unroll
      for (int n = 0; n < 9; ++n) sim_s[w][n] = e[n] * inv;
    }
    lamacc_s[w] = 0.0f;
  }
  __syncthreads();

  // ------------- Phase B: Fw + lambda logit (L2-resident re-reads) -------------
  const int w   = tid & 63;
  const int ci0 = tid >> 6;
  const int wl  = (w > 0) ? w - 1 : 0;
  const int wr  = (w < 63) ? w + 1 : 63;
  float a[9];
  #pragma unroll
  for (int n = 0; n < 9; ++n) a[n] = sim_s[w][n];

  float partial = 0.0f;
  for (int c = ci0; c < CCH; c += 4) {
    size_t base = (size_t)(b * CCH + c) * (HH * WW);
    const float* r0 = Fwp + base + (size_t)hr0 * WW;
    const float* r1 = Fwp + base + (size_t)hr1 * WW;
    const float* r2 = Fwp + base + (size_t)hr2 * WW;
    float fwv = a[0]*r0[wl] + a[1]*r0[w] + a[2]*r0[wr]
              + a[3]*r1[wl] + a[4]*r1[w] + a[5]*r1[wr]
              + a[6]*r2[wl] + a[7]*r2[w] + a[8]*r2[wr];
    size_t idx = base + (size_t)h * WW + w;
    float ftv = Ft[idx];
    out[idx] = fwv;                                  // stage Fw in d_out
    partial += cw[c] * fwv + cw[CCH + c] * ftv;      // lambda logit partial
  }
  atomicAdd(&lamacc_s[w], partial);                  // ds_add_f32
  __syncthreads();
  if (tid < WW)
    lam_s[tid] = 1.0f / (1.0f + expf(-(lamacc_s[tid] + cb[0])));
  __syncthreads();

  // ------------- Phase C: blend -------------
  const float lam = lam_s[w];
  for (int c = ci0; c < CCH; c += 4) {
    size_t idx = (size_t)(b * CCH + c) * (HH * WW) + (size_t)h * WW + w;
    float fwv = out[idx];       // staged Fw (same thread wrote it)
    float ftv = Ft[idx];
    out[idx] = lam * ftv + (1.0f - lam) * fwv;
  }
}

extern "C" void kernel_launch(void* const* d_in, const int* in_sizes, int n_in,
                              void* d_out, int out_size, void* d_ws, size_t ws_size,
                              hipStream_t stream) {
  (void)in_sizes; (void)n_in; (void)out_size; (void)d_ws; (void)ws_size;
  const float* Ft   = (const float*)d_in[0];
  const float* Fwp  = (const float*)d_in[1];
  const float* mask = (const float*)d_in[2];
  const float* cw   = (const float*)d_in[3];
  const float* cb   = (const float*)d_in[4];
  nca_row_kernel<<<BB * HH, 256, 0, stream>>>(Ft, Fwp, mask, cw, cb, (float*)d_out);
}
